// PseudoOneHotEncoding_9414568312899
// MI455X (gfx1250) — matrix operand order repaired and checked by reading, not compiled, not run on hardware
//
#include <hip/hip_runtime.h>
#include <hip/hip_bf16.h>

// DeepFam pseudo-one-hot encoding as a WMMA matmul on gfx1250 (wave32).
//
//   out[N, 21] = onehot(seq)[N, 27(pad 32)] @ table[27(pad 32), 21]
//
// Exact in f16/f32 (table entries are {0, 0.5, 1.0}). Roofline: ~92 MiB of
// traffic at 23.3 TB/s => ~4 us; purely store-bound. WMMA does the per-16-
// token "gather" with the table held in VGPRs (B operand, built once per
// wave). Results are staged per-wave in LDS so the 84 MiB output is written
// as full-wave, 256 B contiguous b64 stores (perfect coalescing) instead of
// strided 64 B / 20 B fragments.

typedef __attribute__((ext_vector_type(16))) _Float16 v16h;
typedef __attribute__((ext_vector_type(8)))  float    v8f;

#define ALPHA 21           // output embedding dim
#define NCLS  27           // vocab classes
#define TILES_PER_GROUP 4  // 64 tokens per group => 1344 floats, 21 b64 waves-stores
#define GROUP_TOK  (TILES_PER_GROUP * 16)
#define GROUP_ELEM (GROUP_TOK * ALPHA)       // 1344 floats per wave group
#define WAVES_PER_BLOCK 8

__global__ __launch_bounds__(256)
void deepfam_onehot_wmma(const int* __restrict__ seq,
                         const float* __restrict__ table,
                         float* __restrict__ out,
                         int ngroups /* = ntok/64 */)
{
    __shared__ float lds[WAVES_PER_BLOCK * GROUP_ELEM];  // 43008 B

    const int lane   = threadIdx.x & 31;
    const int wib    = threadIdx.x >> 5;                 // wave in block
    const int wave   = (int)((blockIdx.x * blockDim.x + threadIdx.x) >> 5);
    const int nwaves = (int)((gridDim.x * blockDim.x) >> 5);

    const int col    = lane & 15;    // N for B/C/D, M for A
    const int hi     = lane >> 4;    // 0 = lanes 0-15, 1 = lanes 16-31
    const int kbaseA = hi << 3;      // A: half-waves interleave K by 8
    const int kbaseB = hi << 4;      // unused; B is built with kbaseA (same K order as A)
    const int mofs   = hi << 3;      // C/D: lanes>=16 hold M = r + 8

    float* const wlds = lds + wib * GROUP_ELEM;          // wave-private region

    const _Float16 one  = (_Float16)1.0f;
    const _Float16 zero = (_Float16)0.0f;

    // ---- Build B (32x16 f16, K x N) once per wave: two N-tiles of padded table.
    // B layout: lane&15 = N, element e -> K = e + (e>=8 ? 8 : 0) + (lane>=16 ? 8 : 0).
    v16h b0, b1;
#pragma unroll
    for (int e = 0; e < 16; ++e) {
        const int K = e + ((e >> 3) << 3) + kbaseA;
        float t0 = 0.0f, t1 = 0.0f;
        if (K < NCLS) {
            t0 = table[K * ALPHA + col];                           // cols 0..15
            if (col < ALPHA - 16) t1 = table[K * ALPHA + 16 + col]; // cols 16..20
        }
        b0[e] = (_Float16)t0;
        b1[e] = (_Float16)t1;
    }

    // ---- Grid-stride over 64-token groups.
    for (int g = wave; g < ngroups; g += nwaves) {
#pragma unroll
        for (int t = 0; t < TILES_PER_GROUP; ++t) {
            const int tok0 = (g * TILES_PER_GROUP + t) << 4;

            // Token class for row M = col (half-waves load the same 16 ints).
            const int c = seq[tok0 + col];

            // A (16x32 f16, M x K) one-hot. A layout: lane&15 = M,
            // element e -> K = e + (e>=8 ? 8 : 0) + (lane>=16 ? 8 : 0).
            v16h a;
#pragma unroll
            for (int e = 0; e < 16; ++e) {
                const int K = e + ((e >> 3) << 3) + kbaseA;
                a[e] = (K == c) ? one : zero;
            }

            v8f acc0 = {}, acc1 = {};
            acc0 = __builtin_amdgcn_wmma_f32_16x16x32_f16(
                false, a, false, b0, (short)0, acc0, false, false);
            acc1 = __builtin_amdgcn_wmma_f32_16x16x32_f16(
                false, a, false, b1, (short)0, acc1, false, false);

            // Scatter C into LDS tile [64 rows x 21 cols] (wave-private, the
            // backend tracks the DScnt RAW hazard for the readback below).
            float* const tl = wlds + (t << 4) * ALPHA;
#pragma unroll
            for (int r = 0; r < 8; ++r)
                tl[(r + mofs) * ALPHA + col] = acc0[r];
            if (col < ALPHA - 16) {              // single divergent region
#pragma unroll
                for (int r = 0; r < 8; ++r)
                    tl[(r + mofs) * ALPHA + 16 + col] = acc1[r];
            }
        }

        // ---- Coalesced writeback: 1344 floats = 21 full-wave b64 stores,
        // each 256 B contiguous and 8 B aligned.
        const float2* __restrict__ s2 = (const float2*)wlds;
        float2* __restrict__ d2 = (float2*)(out + (size_t)g * GROUP_ELEM);
#pragma unroll
        for (int i = 0; i < GROUP_ELEM / 2 / 32; ++i)    // 21 iterations
            d2[i * 32 + lane] = s2[i * 32 + lane];
    }
}

// Scalar tail for ntok % 64 leftover tokens (not hit for the 128x8192 shape).
__global__ __launch_bounds__(256)
void deepfam_onehot_tail(const int* __restrict__ seq,
                         const float* __restrict__ table,
                         float* __restrict__ out,
                         int tok_start, int ntok)
{
    int i = (int)(blockIdx.x * blockDim.x + threadIdx.x);
    int nelem = (ntok - tok_start) * ALPHA;
    if (i >= nelem) return;
    int tok = tok_start + i / ALPHA;
    int c   = i % ALPHA;
    out[tok * ALPHA + c] = table[seq[tok] * ALPHA + c];
}

extern "C" void kernel_launch(void* const* d_in, const int* in_sizes, int n_in,
                              void* d_out, int out_size, void* d_ws, size_t ws_size,
                              hipStream_t stream) {
    const int*   seq   = (const int*)d_in[0];    // [128*8192] int32 classes 0..26
    const float* table = (const float*)d_in[1];  // [27*21] f32
    float*       out   = (float*)d_out;          // [128*8192*21] f32

    const int ntok    = in_sizes[0];
    const int ngroups = ntok / GROUP_TOK;

    if (ngroups > 0) {
        // 8 waves/block; ~1-2 groups per wave at this problem size (16384 groups).
        int blocks = 2048;
        int max_needed = (ngroups + WAVES_PER_BLOCK - 1) / WAVES_PER_BLOCK;
        if (blocks > max_needed) blocks = max_needed;
        if (blocks < 1) blocks = 1;
        deepfam_onehot_wmma<<<blocks, 256, 0, stream>>>(seq, table, out, ngroups);
    }

    const int rem = ntok - ngroups * GROUP_TOK;
    if (rem > 0) {
        const int tok_start = ngroups * GROUP_TOK;
        const int nelem = rem * ALPHA;
        deepfam_onehot_tail<<<(nelem + 255) / 256, 256, 0, stream>>>(
            seq, table, out, tok_start, ntok);
    }
}
